// DenseGATGenerator_13709535609675
// MI455X (gfx1250) — matrix operand order repaired and checked by reading, not compiled, not run on hardware
//
#include <hip/hip_runtime.h>
#include <hip/hip_bf16.h>
#include <math.h>

typedef unsigned short u16;
typedef unsigned int   u32;
typedef long long      i64;

static constexpr int Bc  = 16;
static constexpr int NLRc = 256;
static constexpr int NHRc = 512;
static constexpr int Dc  = 512;
static constexpr int NHc = 8;
static constexpr int Lc  = 4;
static constexpr int Kc  = 4;
static constexpr int FFc = 2048;
static constexpr int HDc = 64;   // head dim

// ---------------- device helpers ----------------

__device__ __forceinline__ float b2f(u16 h) {
    return __uint_as_float(((u32)h) << 16);
}
__device__ __forceinline__ u16 f2b(float f) {
    u32 u = __float_as_uint(f);
    u32 r = u + 0x7FFFu + ((u >> 16) & 1u);   // round-to-nearest-even
    return (u16)(r >> 16);
}
__device__ __forceinline__ float gelu_f(float x) {
    return 0.5f * x * (1.0f + erff(x * 0.70710678118654752440f));
}

// ---------------- WMMA GEMM ----------------
// C[z] = epilogue( alpha * A[z](MxK) @ Bt[z](NxK)^T )
// A, Bt are bf16 row-major. Bt is the transposed B (N rows of K).
// Per-wave 64x32 tile (4x2 fragments -> 8 WMMA per 12 b128 loads per K-step).
// Block = 128 threads = 4 waves in 2x2 -> 128x64 block tile.
// M multiple of 128, N multiple of 64, K multiple of 32 (guaranteed here).

typedef __attribute__((ext_vector_type(16))) __bf16 v16bf;
typedef __attribute__((ext_vector_type(8)))  float  v8f;

union Frag {
    v16bf v;
    uint4 q[2];
};

__device__ __forceinline__ v8f wmma_bf16(const Frag& a, const Frag& b, v8f c) {
    return __builtin_amdgcn_wmma_f32_16x16x32_bf16(
        false, a.v, false, b.v, (short)0, c, false, false);
}

__global__ __launch_bounds__(128)
void gemm_wmma(const u16* __restrict__ A, const u16* __restrict__ Bt,
               float* __restrict__ Cf, u16* __restrict__ Cb,
               const float* __restrict__ bias, const float* __restrict__ Res,
               int Kd, int lda, int ldb, int ldc,
               i64 sA, i64 sB, i64 sCo, i64 sCi, int zdC,
               float alpha, int dogelu)
{
    const int z = blockIdx.z;
    A  += (i64)z * sA;
    Bt += (i64)z * sB;
    const i64 offC = (i64)(z / zdC) * sCo + (i64)(z % zdC) * sCi;

    const int wave = threadIdx.x >> 5;
    const int lane = threadIdx.x & 31;
    const int half = lane >> 4;      // 0: lanes 0-15, 1: lanes 16-31
    const int r    = lane & 15;

    const int m0 = blockIdx.y * 128 + (wave >> 1) * 64;
    const int n0 = blockIdx.x * 64  + (wave & 1) * 32;

    v8f acc[4][2];
    #pragma unroll
    for (int i = 0; i < 4; ++i)
        #pragma unroll
        for (int j = 0; j < 2; ++j)
            acc[i][j] = (v8f){0,0,0,0,0,0,0,0};

    for (int k = 0; k < Kd; k += 32) {
        Frag a[4], b[2];
        // A fragment: lane-half 0 holds K k..k+7 and k+16..k+23;
        //             lane-half 1 holds K k+8..k+15 and k+24..k+31.
        #pragma unroll
        for (int i = 0; i < 4; ++i) {
            const u16* pa = A + (i64)(m0 + i * 16 + r) * lda + k + half * 8;
            a[i].q[0] = *reinterpret_cast<const uint4*>(pa);
            a[i].q[1] = *reinterpret_cast<const uint4*>(pa + 16);
        }
        // B fragment: same K layout as A (half 0: k..k+7, k+16..k+23; half 1: k+8..k+15, k+24..k+31).
        #pragma unroll
        for (int j = 0; j < 2; ++j) {
            const u16* pb = Bt + (i64)(n0 + j * 16 + r) * ldb + k + half * 8;
            b[j].q[0] = *reinterpret_cast<const uint4*>(pb);
            b[j].q[1] = *reinterpret_cast<const uint4*>(pb + 16);
        }
        #pragma unroll
        for (int i = 0; i < 4; ++i)
            #pragma unroll
            for (int j = 0; j < 2; ++j)
                acc[i][j] = wmma_bf16(a[i], b[j], acc[i][j]);
    }

    #pragma unroll
    for (int i = 0; i < 4; ++i) {
        #pragma unroll
        for (int j = 0; j < 2; ++j) {
            const int mt = m0 + i * 16;
            const int nt = n0 + j * 16;
            const int col = nt + r;
            float bv = bias ? bias[col] : 0.0f;
            #pragma unroll
            for (int v = 0; v < 8; ++v) {
                const int row = mt + half * 8 + v;  // C layout: vgpr v -> M=v (+8 hi half)
                float x = alpha * acc[i][j][v] + bv;
                if (dogelu) x = gelu_f(x);
                const i64 ci = offC + (i64)row * ldc + col;
                if (Res) x += Res[ci];
                if (Cf)  Cf[ci] = x;
                if (Cb)  Cb[ci] = f2b(x);
            }
        }
    }
}

// ---------------- LayerNorm (one wave per row, D=512) ----------------

__global__ __launch_bounds__(32)
void ln_rows(const float* __restrict__ X, const float* __restrict__ g,
             const float* __restrict__ bta,
             float* __restrict__ Yf, u16* __restrict__ Yb,
             int Dn, int dogelu)
{
    const int row  = blockIdx.x;
    const int lane = threadIdx.x;
    const int per  = Dn >> 5;            // 16 for D=512
    const float* x = X + (i64)row * Dn;

    float v[16];
    float s = 0.f, sq = 0.f;
    for (int i = 0; i < per; ++i) {
        float t = x[i * 32 + lane];
        v[i] = t; s += t; sq += t * t;
    }
    for (int m = 16; m; m >>= 1) { s += __shfl_xor(s, m, 32); sq += __shfl_xor(sq, m, 32); }
    const float mean = s / (float)Dn;
    const float var  = sq / (float)Dn - mean * mean;
    const float inv  = rsqrtf(var + 1e-5f);
    for (int i = 0; i < per; ++i) {
        const int c = i * 32 + lane;
        float y = (v[i] - mean) * inv * g[c] + bta[c];
        if (dogelu) y = gelu_f(y);
        if (Yf) Yf[(i64)row * Dn + c] = y;
        if (Yb) Yb[(i64)row * Dn + c] = f2b(y);
    }
}

// ---------------- Softmax over rows of bf16 scores (in-place) ----------------
// Optional bias: ebs[0] * A[b, n, m] * ebw[h]  (encoder edge bias)

__global__ __launch_bounds__(32)
void softmax_rows(u16* __restrict__ S, int Nc, int Nseq,
                  const float* __restrict__ Ab,
                  const float* __restrict__ ebs,
                  const float* __restrict__ ebw)
{
    const int row  = blockIdx.x;
    const int lane = threadIdx.x;
    const int per  = Nc >> 5;            // 8 or 16
    u16* p = S + (i64)row * Nc;

    float scale = 0.f;
    const float* arow = nullptr;
    if (Ab) {
        const int n = row % Nseq;
        const int h = (row / Nseq) % NHc;
        const int b = row / (Nseq * NHc);
        scale = ebs[0] * ebw[h];
        arow  = Ab + ((i64)b * Nseq + n) * Nc;
    }

    float v[16];
    float mx = -1e30f;
    for (int i = 0; i < per; ++i) {
        const int c = i * 32 + lane;
        float x = b2f(p[c]);
        if (arow) x += scale * arow[c];
        v[i] = x; mx = fmaxf(mx, x);
    }
    for (int m = 16; m; m >>= 1) mx = fmaxf(mx, __shfl_xor(mx, m, 32));
    float sum = 0.f;
    for (int i = 0; i < per; ++i) { v[i] = expf(v[i] - mx); sum += v[i]; }
    for (int m = 16; m; m >>= 1) sum += __shfl_xor(sum, m, 32);
    const float inv = 1.0f / sum;
    for (int i = 0; i < per; ++i) p[i * 32 + lane] = f2b(v[i] * inv);
}

// ---------------- layout / conversion kernels ----------------

__global__ void cvt_f32_bf16(const float* __restrict__ in, u16* __restrict__ out, i64 n) {
    const i64 t = (i64)blockIdx.x * 256 + threadIdx.x;
    if (t < n) out[t] = f2b(in[t]);
}

__global__ void tr_f32_bf16(const float* __restrict__ in, u16* __restrict__ out, int R, int C) {
    const i64 z = blockIdx.y;
    const i64 t = (i64)blockIdx.x * 256 + threadIdx.x;
    if (t >= (i64)R * C) return;
    const int rr = (int)(t / C), cc = (int)(t % C);
    out[z * (i64)R * C + (i64)cc * R + rr] = f2b(in[z * (i64)R * C + t]);
}

__global__ void tr_bf16(const u16* __restrict__ in, u16* __restrict__ out, int R, int C) {
    const i64 z = blockIdx.y;
    const i64 t = (i64)blockIdx.x * 256 + threadIdx.x;
    if (t >= (i64)R * C) return;
    const int rr = (int)(t / C), cc = (int)(t % C);
    out[z * (i64)R * C + (i64)cc * R + rr] = in[z * (i64)R * C + t];
}

__global__ void tr_f32(const float* __restrict__ in, float* __restrict__ out, int R, int C) {
    const i64 z = blockIdx.y;
    const i64 t = (i64)blockIdx.x * 256 + threadIdx.x;
    if (t >= (i64)R * C) return;
    const int rr = (int)(t / C), cc = (int)(t % C);
    out[z * (i64)R * C + (i64)cc * R + rr] = in[z * (i64)R * C + t];
}

// split fused QKV (B,N,3,NH,hd) into Q,K (B,NH,N,hd) and V^T (B,NH,hd,N), all bf16
__global__ void qkv_split(const u16* __restrict__ QKV,
                          u16* __restrict__ Q, u16* __restrict__ Kx, u16* __restrict__ Vt,
                          int Nseq)
{
    const i64 t = (i64)blockIdx.x * 256 + threadIdx.x;
    const i64 tot = (i64)Bc * Nseq * Dc;
    if (t >= tot) return;
    const int e = (int)(t % HDc);
    const int h = (int)((t / HDc) % NHc);
    const int n = (int)((t / Dc) % Nseq);
    const int b = (int)(t / ((i64)Dc * Nseq));
    const i64 base = ((i64)(b * Nseq + n) * 3) * Dc + h * HDc + e;
    const i64 qk   = ((i64)(b * NHc + h) * Nseq + n) * HDc + e;
    Q[qk]  = QKV[base];
    Kx[qk] = QKV[base + Dc];
    Vt[((i64)(b * NHc + h) * HDc + e) * Nseq + n] = QKV[base + 2 * Dc];
}

// WsumT[e,d] = sum_k dec_W[k,d,e]  (bf16, ready as B^T for G = H @ Wsum)
__global__ void decw_sum(const float* __restrict__ W, u16* __restrict__ out) {
    const i64 t = (i64)blockIdx.x * 256 + threadIdx.x;
    if (t >= (i64)Dc * Dc) return;
    const int e = (int)(t / Dc), d = (int)(t % Dc);
    float s = 0.f;
    for (int k = 0; k < Kc; ++k) s += W[((i64)k * Dc + d) * Dc + e];
    out[(i64)e * Dc + d] = f2b(s);
}

// out[b, triu(i,j)] = softplus(0.5*(P[b,i,j]+P[b,j,i]) + db)
__global__ void decode_out(const float* __restrict__ P, const float* __restrict__ db,
                           float* __restrict__ out, int Nn, i64 T)
{
    const i64 t = (i64)blockIdx.x * 256 + threadIdx.x;
    const i64 tot = (i64)Bc * Nn * Nn;
    if (t >= tot) return;
    const int c  = (int)(t % Nn);
    const int rr = (int)((t / Nn) % Nn);
    const int b  = (int)(t / ((i64)Nn * Nn));
    if (c <= rr) return;
    const float x = 0.5f * (P[((i64)b * Nn + rr) * Nn + c] + P[((i64)b * Nn + c) * Nn + rr]) + db[0];
    const float y = (x > 20.f) ? x : log1pf(expf(x));
    const i64 idx = (i64)rr * (Nn - 1) - (i64)rr * (rr - 1) / 2 + (c - rr - 1);
    out[(i64)b * T + idx] = y;
}

// ---------------- host-side helpers ----------------

static inline void launch_gemm(hipStream_t s,
                               const u16* A, const u16* Bt, float* Cf, u16* Cb,
                               const float* bias, const float* Res,
                               int M, int N, int K, int lda, int ldb, int ldc,
                               int batch, i64 sA, i64 sB, i64 sCo, i64 sCi, int zdC,
                               float alpha, int dogelu)
{
    dim3 grid(N / 64, M / 128, batch), block(128);
    gemm_wmma<<<grid, block, 0, s>>>(A, Bt, Cf, Cb, bias, Res, K, lda, ldb, ldc,
                                     sA, sB, sCo, sCi, zdC, alpha, dogelu);
}

struct Bufs {
    float* Hf;  u16* Hnb; u16* QKVb;
    u16* Qb; u16* Kb; u16* Vtb; u16* Sb; u16* Ob; u16* F1b;
};

// One pre-LN transformer block: H += attn(LN1(H)); H += ffn(LN2(H))
static void run_block(hipStream_t s, const Bufs& w, int Nseq,
                      const float* n1g, const float* n1b,
                      const u16* qkvWt, const float* qkvb,
                      const u16* projWt, const float* projb,
                      const float* n2g, const float* n2b,
                      const u16* f1Wt, const float* f1bp,
                      const u16* f2Wt, const float* f2bp,
                      const float* Ab, const float* ebs, const float* ebw)
{
    const int rows = Bc * Nseq;
    // LN1 -> bf16
    ln_rows<<<rows, 32, 0, s>>>(w.Hf, n1g, n1b, nullptr, w.Hnb, Dc, 0);
    // QKV
    launch_gemm(s, w.Hnb, qkvWt, nullptr, w.QKVb, qkvb, nullptr,
                rows, 3 * Dc, Dc, Dc, Dc, 3 * Dc, 1, 0, 0, 0, 0, 1, 1.0f, 0);
    qkv_split<<<(int)(((i64)rows * Dc + 255) / 256), 256, 0, s>>>(w.QKVb, w.Qb, w.Kb, w.Vtb, Nseq);
    // scores = (1/sqrt(hd)) * Q @ K^T   (per (b,h))
    launch_gemm(s, w.Qb, w.Kb, nullptr, w.Sb, nullptr, nullptr,
                Nseq, Nseq, HDc, HDc, HDc, Nseq,
                Bc * NHc, (i64)Nseq * HDc, (i64)Nseq * HDc,
                (i64)Nseq * Nseq, 0, 1, 0.125f, 0);
    // softmax (+ optional edge bias)
    softmax_rows<<<Bc * NHc * Nseq, 32, 0, s>>>(w.Sb, Nseq, Nseq, Ab, ebs, ebw);
    // O = probs @ V  -> written merged as (b, n, h*hd)
    launch_gemm(s, w.Sb, w.Vtb, nullptr, w.Ob, nullptr, nullptr,
                Nseq, HDc, Nseq, Nseq, Nseq, Dc,
                Bc * NHc, (i64)Nseq * Nseq, (i64)HDc * Nseq,
                (i64)Nseq * Dc, HDc, NHc, 1.0f, 0);
    // proj + residual
    launch_gemm(s, w.Ob, projWt, w.Hf, nullptr, projb, w.Hf,
                rows, Dc, Dc, Dc, Dc, Dc, 1, 0, 0, 0, 0, 1, 1.0f, 0);
    // LN2 -> bf16
    ln_rows<<<rows, 32, 0, s>>>(w.Hf, n2g, n2b, nullptr, w.Hnb, Dc, 0);
    // FFN
    launch_gemm(s, w.Hnb, f1Wt, nullptr, w.F1b, f1bp, nullptr,
                rows, FFc, Dc, Dc, Dc, FFc, 1, 0, 0, 0, 0, 1, 1.0f, 1);
    launch_gemm(s, w.F1b, f2Wt, w.Hf, nullptr, f2bp, w.Hf,
                rows, Dc, FFc, FFc, FFc, Dc, 1, 0, 0, 0, 0, 1, 1.0f, 0);
}

// ---------------- entry point ----------------

extern "C" void kernel_launch(void* const* d_in, const int* in_sizes, int n_in,
                              void* d_out, int out_size, void* d_ws, size_t ws_size,
                              hipStream_t stream)
{
    const float* A_lr   = (const float*)d_in[0];
    const float* X_lr   = (const float*)d_in[1];
    const float* ip_W   = (const float*)d_in[2];
    const float* ip_b   = (const float*)d_in[3];
    const float* ip_g   = (const float*)d_in[4];
    const float* ip_bt  = (const float*)d_in[5];
    const float* e_n1g  = (const float*)d_in[6];
    const float* e_n1b  = (const float*)d_in[7];
    const float* e_qkvW = (const float*)d_in[8];
    const float* e_qkvb = (const float*)d_in[9];
    const float* e_projW= (const float*)d_in[10];
    const float* e_projb= (const float*)d_in[11];
    const float* e_ebW  = (const float*)d_in[12];
    const float* e_ebs  = (const float*)d_in[13];
    const float* e_n2g  = (const float*)d_in[14];
    const float* e_n2b  = (const float*)d_in[15];
    const float* e_f1W  = (const float*)d_in[16];
    const float* e_f1b  = (const float*)d_in[17];
    const float* e_f2W  = (const float*)d_in[18];
    const float* e_f2b  = (const float*)d_in[19];
    const float* encn_g = (const float*)d_in[20];
    const float* encn_b = (const float*)d_in[21];
    const float* up1W   = (const float*)d_in[22];
    const float* up1b   = (const float*)d_in[23];
    const float* up2W   = (const float*)d_in[24];
    const float* up2b   = (const float*)d_in[25];
    const float* r_n1g  = (const float*)d_in[26];
    const float* r_n1b  = (const float*)d_in[27];
    const float* r_qkvW = (const float*)d_in[28];
    const float* r_qkvb = (const float*)d_in[29];
    const float* r_projW= (const float*)d_in[30];
    const float* r_projb= (const float*)d_in[31];
    const float* r_n2g  = (const float*)d_in[32];
    const float* r_n2b  = (const float*)d_in[33];
    const float* r_f1W  = (const float*)d_in[34];
    const float* r_f1b  = (const float*)d_in[35];
    const float* r_f2W  = (const float*)d_in[36];
    const float* r_f2b  = (const float*)d_in[37];
    const float* hrn_g  = (const float*)d_in[38];
    const float* hrn_b  = (const float*)d_in[39];
    const float* dec_W  = (const float*)d_in[40];
    const float* dec_b  = (const float*)d_in[41];
    (void)in_sizes; (void)n_in; (void)out_size;

    // ---- bump allocator over workspace ----
    char* ws = (char*)d_ws;
    size_t off = 0;
    auto alloc = [&](size_t bytes) -> void* {
        void* p = ws + off;
        off += (bytes + 255) & ~(size_t)255;
        return p;
    };

    float* Hf   = (float*)alloc((size_t)Bc * NHRc * Dc * 4);
    u16* Hnb    = (u16*)  alloc((size_t)Bc * NHRc * Dc * 2);
    u16* Xb     = (u16*)  alloc((size_t)Bc * NLRc * NLRc * 2);
    u16* QKVb   = (u16*)  alloc((size_t)Bc * NHRc * 3 * Dc * 2);
    u16* Qb     = (u16*)  alloc((size_t)Bc * NHRc * Dc * 2);
    u16* Kb     = (u16*)  alloc((size_t)Bc * NHRc * Dc * 2);
    u16* Vtb    = (u16*)  alloc((size_t)Bc * NHRc * Dc * 2);
    u16* Sb     = (u16*)  alloc((size_t)Bc * NHc * NHRc * NHRc * 2);
    u16* Ob     = (u16*)  alloc((size_t)Bc * NHRc * Dc * 2);
    u16* F1b    = (u16*)  alloc((size_t)Bc * NHRc * FFc * 2);
    u16* TrA    = (u16*)  alloc((size_t)Bc * Dc * NHRc * 2);
    u16* U1b    = (u16*)  alloc((size_t)Bc * Dc * NHRc * 2);
    float* Ht2  = (float*)alloc((size_t)Bc * Dc * NHRc * 4);
    u16* Gb     = (u16*)  alloc((size_t)Bc * NHRc * Dc * 2);
    float* Pf   = (float*)alloc((size_t)Bc * NHRc * NHRc * 4);
    // transposed bf16 weights
    u16* ipWt   = (u16*)alloc((size_t)Dc * NLRc * 2);
    u16* qkvWt  = (u16*)alloc((size_t)Lc * 3 * Dc * Dc * 2);
    u16* projWt = (u16*)alloc((size_t)Lc * Dc * Dc * 2);
    u16* f1Wt   = (u16*)alloc((size_t)Lc * FFc * Dc * 2);
    u16* f2Wt   = (u16*)alloc((size_t)Lc * Dc * FFc * 2);
    u16* up1Wt  = (u16*)alloc((size_t)NHRc * NLRc * 2);
    u16* up2Wt  = (u16*)alloc((size_t)NHRc * NHRc * 2);
    u16* rqkvWt = (u16*)alloc((size_t)3 * Dc * Dc * 2);
    u16* rprojWt= (u16*)alloc((size_t)Dc * Dc * 2);
    u16* rf1Wt  = (u16*)alloc((size_t)FFc * Dc * 2);
    u16* rf2Wt  = (u16*)alloc((size_t)Dc * FFc * 2);
    u16* WsumT  = (u16*)alloc((size_t)Dc * Dc * 2);
    if (off > ws_size) return;   // workspace too small; bail deterministically

    hipStream_t s = stream;
    auto trg = [](int R, int C) { return (int)(((i64)R * C + 255) / 256); };

    // ---- weight prep: f32 -> bf16 transposed ----
    tr_f32_bf16<<<dim3(trg(NLRc, Dc), 1),    256, 0, s>>>(ip_W,   ipWt,   NLRc, Dc);
    tr_f32_bf16<<<dim3(trg(Dc, 3 * Dc), Lc), 256, 0, s>>>(e_qkvW, qkvWt,  Dc, 3 * Dc);
    tr_f32_bf16<<<dim3(trg(Dc, Dc), Lc),     256, 0, s>>>(e_projW,projWt, Dc, Dc);
    tr_f32_bf16<<<dim3(trg(Dc, FFc), Lc),    256, 0, s>>>(e_f1W,  f1Wt,   Dc, FFc);
    tr_f32_bf16<<<dim3(trg(FFc, Dc), Lc),    256, 0, s>>>(e_f2W,  f2Wt,   FFc, Dc);
    tr_f32_bf16<<<dim3(trg(NLRc, NHRc), 1),  256, 0, s>>>(up1W,   up1Wt,  NLRc, NHRc);
    tr_f32_bf16<<<dim3(trg(NHRc, NHRc), 1),  256, 0, s>>>(up2W,   up2Wt,  NHRc, NHRc);
    tr_f32_bf16<<<dim3(trg(Dc, 3 * Dc), 1),  256, 0, s>>>(r_qkvW, rqkvWt, Dc, 3 * Dc);
    tr_f32_bf16<<<dim3(trg(Dc, Dc), 1),      256, 0, s>>>(r_projW,rprojWt,Dc, Dc);
    tr_f32_bf16<<<dim3(trg(Dc, FFc), 1),     256, 0, s>>>(r_f1W,  rf1Wt,  Dc, FFc);
    tr_f32_bf16<<<dim3(trg(FFc, Dc), 1),     256, 0, s>>>(r_f2W,  rf2Wt,  FFc, Dc);
    decw_sum<<<trg(Dc, Dc), 256, 0, s>>>(dec_W, WsumT);

    // ---- input projection: H = gelu(LN(X_lr @ ip_W + ip_b)) ----
    cvt_f32_bf16<<<trg(Bc * NLRc, NLRc), 256, 0, s>>>(X_lr, Xb, (i64)Bc * NLRc * NLRc);
    launch_gemm(s, Xb, ipWt, Hf, nullptr, ip_b, nullptr,
                Bc * NLRc, Dc, NLRc, NLRc, NLRc, Dc, 1, 0, 0, 0, 0, 1, 1.0f, 0);
    ln_rows<<<Bc * NLRc, 32, 0, s>>>(Hf, ip_g, ip_bt, Hf, nullptr, Dc, 1);

    Bufs w{Hf, Hnb, QKVb, Qb, Kb, Vtb, Sb, Ob, F1b};

    // ---- 4 encoder layers @ N=256 with edge bias ----
    for (int l = 0; l < Lc; ++l) {
        run_block(s, w, NLRc,
                  e_n1g + (i64)l * Dc, e_n1b + (i64)l * Dc,
                  qkvWt + (i64)l * 3 * Dc * Dc, e_qkvb + (i64)l * 3 * Dc,
                  projWt + (i64)l * Dc * Dc, e_projb + (i64)l * Dc,
                  e_n2g + (i64)l * Dc, e_n2b + (i64)l * Dc,
                  f1Wt + (i64)l * FFc * Dc, e_f1b + (i64)l * FFc,
                  f2Wt + (i64)l * Dc * FFc, e_f2b + (i64)l * Dc,
                  A_lr, e_ebs + l, e_ebW + (i64)l * NHc);
    }

    // ---- final encoder LN, upsample along sequence dim ----
    ln_rows<<<Bc * NLRc, 32, 0, s>>>(Hf, encn_g, encn_b, nullptr, Hnb, Dc, 0);
    tr_bf16<<<dim3(trg(NLRc, Dc), Bc), 256, 0, s>>>(Hnb, TrA, NLRc, Dc);   // (B, D, N_LR)
    launch_gemm(s, TrA, up1Wt, nullptr, U1b, up1b, nullptr,
                Dc, NHRc, NLRc, NLRc, NLRc, NHRc,
                Bc, (i64)Dc * NLRc, 0, (i64)Dc * NHRc, 0, 1, 1.0f, 1);
    launch_gemm(s, U1b, up2Wt, Ht2, nullptr, up2b, nullptr,
                Dc, NHRc, NHRc, NHRc, NHRc, NHRc,
                Bc, (i64)Dc * NHRc, 0, (i64)Dc * NHRc, 0, 1, 1.0f, 0);
    tr_f32<<<dim3(trg(Dc, NHRc), Bc), 256, 0, s>>>(Ht2, Hf, Dc, NHRc);     // back to (B, N_HR, D)

    // ---- refinement block @ N=512, no attention bias ----
    run_block(s, w, NHRc,
              r_n1g, r_n1b, rqkvWt, r_qkvb, rprojWt, r_projb,
              r_n2g, r_n2b, rf1Wt, r_f1b, rf2Wt, r_f2b,
              nullptr, nullptr, nullptr);

    // ---- decoder: P = (H @ sum_k W_k) @ H^T / K ----
    ln_rows<<<Bc * NHRc, 32, 0, s>>>(Hf, hrn_g, hrn_b, nullptr, Hnb, Dc, 0);
    launch_gemm(s, Hnb, WsumT, nullptr, Gb, nullptr, nullptr,
                Bc * NHRc, Dc, Dc, Dc, Dc, Dc, 1, 0, 0, 0, 0, 1, 1.0f, 0);
    launch_gemm(s, Gb, Hnb, Pf, nullptr, nullptr, nullptr,
                NHRc, NHRc, Dc, Dc, Dc, NHRc,
                Bc, (i64)NHRc * Dc, (i64)NHRc * Dc,
                (i64)NHRc * NHRc, 0, 1, 1.0f / (float)Kc, 0);

    // ---- symmetrize + upper-triangle + softplus ----
    const i64 T = (i64)NHRc * (NHRc - 1) / 2;
    decode_out<<<trg(Bc * NHRc, NHRc), 256, 0, s>>>(Pf, dec_b, (float*)d_out, NHRc, T);
}
